// GCRINT_41755672052276
// MI455X (gfx1250) — matrix operand order repaired and checked by reading, not compiled, not run on hardware
//
#include <hip/hip_runtime.h>
#include <cstdint>

// ---------------------------------------------------------------------------
// Problem constants
// ---------------------------------------------------------------------------
#define BB 16
#define TT 62
#define NN 529
#define CC 32
#define HH 64
#define G4 256           // 4*H gate width
#define MSEQ (BB * NN)   // 8464 sequences (multiple of 16)
#define NPAD 544         // 529 padded to multiple of 16
#define MROW 1024        // B*H rows of gsum

typedef __attribute__((ext_vector_type(16))) _Float16 v16h;
typedef __attribute__((ext_vector_type(8)))  float    v8f;

// ---------------------------------------------------------------------------
// WMMA fragment helpers (CDNA5 gfx1250, wave32, 16x16x32 f16)
// A layout (ISA 7.12.2, 16-bit A 16x32): lane<16 -> M=lane, K in {0..7,16..23};
// lane>=16 -> M=lane-16, K offset +8. Element 2p+h lives in VGPR p half h.
// ---------------------------------------------------------------------------
__device__ __forceinline__ int lane_id() { return (int)(threadIdx.x & 31u); }

__device__ __forceinline__ v16h load_a_frag(const _Float16* A, int lda) {
  const int lane = lane_id();
  const _Float16* p = A + (lane & 15) * lda + ((lane >> 4) << 3);
  v16h f;
#pragma unroll
  for (int pr = 0; pr < 8; ++pr) {
    const int kk = ((pr & 3) << 1) + ((pr >> 2) << 4);  // 0,2,4,6,16,18,20,22
    f[2 * pr]     = p[kk];
    f[2 * pr + 1] = p[kk + 1];
  }
  return f;
}

// B is K x N (32x16 tile), row-major with leading dim ldb.
// lane<16 -> N=lane, K in {0..7,16..23} ; lane>=16 -> N=lane-16, K offset +8.
__device__ __forceinline__ v16h load_b_frag(const _Float16* Bm, int ldb) {
  const int lane = lane_id();
  const _Float16* p = Bm + (lane & 15) + ((lane >> 4) << 3) * ldb;
  v16h f;
#pragma unroll
  for (int e = 0; e < 16; ++e) f[e] = p[((e & 7) + ((e >> 3) << 4)) * ldb];
  return f;
}

__device__ __forceinline__ float sigmoidf_(float x) {
  return 1.0f / (1.0f + __expf(-x));
}

// ---------------------------------------------------------------------------
// Kernel 1: tiny precompute (one block).
//  - rank-2 collapsed input projection vectors per direction: A,B,K (256 each)
//  - whh^T staged as f16 [64][256] (WMMA B operand)
//  - Wcomb = end2_w @ end1_w @ gconv_w   [62][192]
//  - bias_comb[o] = slin*(end2@(end1@gconv_b+e1b)+e2b)[o] + lin_b
// ---------------------------------------------------------------------------
__global__ __launch_bounds__(256) void prep_kernel(
    const float* sw, const float* sb, const float* cw, const float* cb,
    const float* wih_fw, const float* whh_fw, const float* bih_fw, const float* bhh_fw,
    const float* wih_bw, const float* whh_bw, const float* bih_bw, const float* bhh_bw,
    const float* gconv_w, const float* gconv_b,
    const float* end1_w, const float* end1_b,
    const float* end2_w, const float* end2_b,
    const float* lin_w, const float* lin_b,
    float* vec_fw, float* vec_bw,
    _Float16* whhT_fw, _Float16* whhT_bw,
    float* E, float* Wcomb, float* biasc) {
  const int tid = threadIdx.x;
  __shared__ float ub[HH];

  // Stage 1: collapsed gate vectors (tid == gate index g, exactly 256)
  {
    float a1 = 0.f, b1 = 0.f, k1 = 0.f, a2 = 0.f, b2 = 0.f, k2 = 0.f;
    for (int c = 0; c < CC; ++c) {
      const float w1 = wih_fw[tid * CC + c];
      const float w2 = wih_bw[tid * CC + c];
      const float bias = sb[c] + cb[c];
      a1 += w1 * sw[c]; b1 += w1 * cw[c]; k1 += w1 * bias;
      a2 += w2 * sw[c]; b2 += w2 * cw[c]; k2 += w2 * bias;
    }
    vec_fw[tid] = a1; vec_fw[256 + tid] = b1;
    vec_fw[512 + tid] = k1 + bih_fw[tid] + bhh_fw[tid];
    vec_bw[tid] = a2; vec_bw[256 + tid] = b2;
    vec_bw[512 + tid] = k2 + bih_bw[tid] + bhh_bw[tid];
  }
  // Stage 2: whh^T as f16 [k][g]
  for (int i = tid; i < HH * G4; i += 256) {
    const int k = i >> 8, g = i & 255;
    whhT_fw[i] = (_Float16)whh_fw[g * HH + k];
    whhT_bw[i] = (_Float16)whh_bw[g * HH + k];
  }
  // Stage 3: E = end2_w @ end1_w   [62][32]
  for (int i = tid; i < TT * CC; i += 256) {
    const int o = i / CC, c = i % CC;
    float s = 0.f;
    for (int h = 0; h < HH; ++h) s += end2_w[o * HH + h] * end1_w[h * CC + c];
    E[i] = s;
  }
  if (tid == 0) {
    float s = 0.f;
    for (int t = 0; t < TT; ++t) s += lin_w[t];
    biasc[62] = s;  // slin
  }
  if (tid < HH) {
    float s = 0.f;
    for (int c = 0; c < CC; ++c) s += end1_w[tid * CC + c] * gconv_b[c];
    ub[tid] = s + end1_b[tid];
  }
  __syncthreads();
  // Stage 4: Wcomb = E @ gconv_w   [62][192]
  for (int i = tid; i < TT * 3 * HH; i += 256) {
    const int o = i / (3 * HH), ii = i % (3 * HH);
    float s = 0.f;
    for (int c = 0; c < CC; ++c) s += E[o * CC + c] * gconv_w[c * (3 * HH) + ii];
    Wcomb[i] = s;
  }
  if (tid < TT) {
    float s = 0.f;
    for (int h = 0; h < HH; ++h) s += end2_w[tid * HH + h] * ub[h];
    biasc[tid] = biasc[62] * (s + end2_b[tid]) + lin_b[0];
  }
}

// ---------------------------------------------------------------------------
// Kernel 2: adaptive adjacency  adp = softmax(relu(nv1@nv2), axis=1)
// stored as zero-padded f16 [544][544] (pad rows/cols = 0 so WMMA K-padding
// is inert).  One block per row.
// ---------------------------------------------------------------------------
__global__ __launch_bounds__(256) void adp_kernel(const float* nv1, const float* nv2,
                                                  _Float16* adp) {
  const int r = blockIdx.x;
  const int tid = threadIdx.x;
  __shared__ float buf[NPAD];
  __shared__ float red[256];
  if (r >= NN) {
    for (int j = tid; j < NPAD; j += 256) adp[(size_t)r * NPAD + j] = (_Float16)0.f;
    return;
  }
  float a[10];
#pragma unroll
  for (int k = 0; k < 10; ++k) a[k] = nv1[r * 10 + k];
  for (int j = tid; j < NPAD; j += 256) {
    float v = -1e30f;
    if (j < NN) {
      float s = 0.f;
#pragma unroll
      for (int k = 0; k < 10; ++k) s += a[k] * nv2[k * NN + j];
      v = fmaxf(s, 0.f);
    }
    buf[j] = v;
  }
  __syncthreads();
  float m = -1e30f;
  for (int j = tid; j < NN; j += 256) m = fmaxf(m, buf[j]);
  red[tid] = m; __syncthreads();
  for (int s = 128; s > 0; s >>= 1) {
    if (tid < s) red[tid] = fmaxf(red[tid], red[tid + s]);
    __syncthreads();
  }
  const float mx = red[0]; __syncthreads();
  float sum = 0.f;
  for (int j = tid; j < NN; j += 256) {
    const float e = __expf(buf[j] - mx);
    buf[j] = e; sum += e;
  }
  red[tid] = sum; __syncthreads();
  for (int s = 128; s > 0; s >>= 1) {
    if (tid < s) red[tid] += red[tid + s];
    __syncthreads();
  }
  const float inv = 1.f / red[0];
  for (int j = tid; j < NPAD; j += 256)
    adp[(size_t)r * NPAD + j] = (j < NN) ? (_Float16)(buf[j] * inv) : (_Float16)0.f;
}

// ---------------------------------------------------------------------------
// Kernel 3: LSTM recurrence. One block = one 16-sequence M-tile, all 62 steps.
// h (f16) and c (f32) live in LDS; whh^T B-fragments stay in VGPRs for the
// whole time loop. Gate pre-activation = WMMA(h, whh^T) + rank-2 input term.
// blockIdx.y selects direction. Hidden sequences stored f16 for the combiner.
// ---------------------------------------------------------------------------
__global__ __launch_bounds__(256) void lstm_kernel(
    const float* it_fw, const float* mk_fw,
    const float* it_bw, const float* mk_bw,
    const _Float16* whhT_fw, const _Float16* whhT_bw,
    const float* vec_fw, const float* vec_bw,
    _Float16* hseq_fw, _Float16* hseq_bw) {
  const float* it; const float* mk; const _Float16* whhT; const float* vec;
  _Float16* hseq;
  if (blockIdx.y == 0) { it = it_fw; mk = mk_fw; whhT = whhT_fw; vec = vec_fw; hseq = hseq_fw; }
  else                 { it = it_bw; mk = mk_bw; whhT = whhT_bw; vec = vec_bw; hseq = hseq_bw; }
  const float* Av = vec;
  const float* Bv = vec + 256;
  const float* Kv = vec + 512;

  __shared__ _Float16 sh_h[16 * HH];
  __shared__ float    sh_c[16 * HH];
  __shared__ float    sh_g[16 * G4];
  __shared__ float    sh_in[32];

  const int tid = threadIdx.x;
  for (int i = tid; i < 16 * HH; i += 256) { sh_h[i] = (_Float16)0.f; sh_c[i] = 0.f; }

  const int wave = tid >> 5;
  const int lane = tid & 31;
  const int nt0 = wave * 2, nt1 = wave * 2 + 1;
  // B fragments resident for all 62 steps (4 WMMA per wave per step).
  const v16h b00 = load_b_frag(whhT + nt0 * 16, G4);
  const v16h b01 = load_b_frag(whhT + 32 * G4 + nt0 * 16, G4);
  const v16h b10 = load_b_frag(whhT + nt1 * 16, G4);
  const v16h b11 = load_b_frag(whhT + 32 * G4 + nt1 * 16, G4);

  const int mbase = (int)blockIdx.x * 16;

  for (int t = 0; t < TT; ++t) {
    if (tid < 16) {
      const int m = mbase + tid;
      const int b = m / NN;
      const int n = m - b * NN;
      const size_t idx = ((size_t)b * TT + t) * NN + n;
      sh_in[tid] = it[idx];
      sh_in[16 + tid] = mk[idx];
    }
    __syncthreads();

    const v16h a0 = load_a_frag(sh_h, HH);       // K = 0..31
    const v16h a1 = load_a_frag(sh_h + 32, HH);  // K = 32..63
    v8f acc0 = {}; v8f acc1 = {};
    acc0 = __builtin_amdgcn_wmma_f32_16x16x32_f16(false, a0, false, b00, (short)0, acc0, false, false);
    acc0 = __builtin_amdgcn_wmma_f32_16x16x32_f16(false, a1, false, b01, (short)0, acc0, false, false);
    acc1 = __builtin_amdgcn_wmma_f32_16x16x32_f16(false, a0, false, b10, (short)0, acc1, false, false);
    acc1 = __builtin_amdgcn_wmma_f32_16x16x32_f16(false, a1, false, b11, (short)0, acc1, false, false);

    const int cn = lane & 15;
    const int row0 = (lane >> 4) << 3;
    const int c0 = nt0 * 16 + cn, c1 = nt1 * 16 + cn;
#pragma unroll
    for (int r = 0; r < 8; ++r) {
      const int mr = row0 + r;
      sh_g[mr * G4 + c0] = acc0[r] + sh_in[mr] * Av[c0] + sh_in[16 + mr] * Bv[c0] + Kv[c0];
      sh_g[mr * G4 + c1] = acc1[r] + sh_in[mr] * Av[c1] + sh_in[16 + mr] * Bv[c1] + Kv[c1];
    }
    __syncthreads();

    for (int i = tid; i < 16 * HH; i += 256) {
      const int s = i >> 6, hh = i & 63;
      const float gi = sh_g[s * G4 + hh];
      const float gf = sh_g[s * G4 + 64 + hh];
      const float gg = sh_g[s * G4 + 128 + hh];
      const float go = sh_g[s * G4 + 192 + hh];
      const float c = sigmoidf_(gf) * sh_c[i] + sigmoidf_(gi) * tanhf(gg);
      sh_c[i] = c;
      const float h = sigmoidf_(go) * tanhf(c);
      sh_h[i] = (_Float16)h;
      hseq[((size_t)(mbase + s) * TT + t) * HH + hh] = (_Float16)h;
    }
    __syncthreads();
  }
}

// ---------------------------------------------------------------------------
// Kernel 4: t-reduction + tanh combine (the whole diffusion operates on this):
//   gsum[b*64+h][n] = sum_t lin_w[t] * tanh((h_fw[m,t,h]+h_bw[m,T-1-t,h])/2)
// zero-padded to [1024][544] f16.
// ---------------------------------------------------------------------------
__global__ __launch_bounds__(256) void combine_kernel(
    const _Float16* hfw, const _Float16* hbw, const float* lin_w, _Float16* gsum) {
  const int idx = (int)(blockIdx.x * 256 + threadIdx.x);
  if (idx >= MROW * NPAD) return;
  const int row = idx / NPAD;
  const int n = idx - row * NPAD;
  _Float16 out = (_Float16)0.f;
  if (n < NN) {
    const int b = row >> 6, h = row & 63;
    const size_t m = (size_t)b * NN + n;
    const _Float16* pf = hfw + (m * TT) * HH + h;
    const _Float16* pb = hbw + (m * TT) * HH + h;
    float s = 0.f;
    for (int t = 0; t < TT; ++t)
      s += lin_w[t] * tanhf(0.5f * ((float)pf[t * HH] + (float)pb[(TT - 1 - t) * HH]));
    out = (_Float16)s;
  }
  gsum[idx] = out;
}

// ---------------------------------------------------------------------------
// Kernel 5: generic f16 WMMA GEMM, C[M][N] = A[M][K] x B[K][N], all dims
// multiples of 16 (padded buffers). 8 waves/block, 1 output tile per wave.
// Used for x1 = gsum@adp and x2 = x1@adp (diffusion over nodes).
// ---------------------------------------------------------------------------
__global__ __launch_bounds__(256) void gemm_f16(const _Float16* A, const _Float16* B,
                                                _Float16* C, int M, int N, int K) {
  const int wave = (int)(threadIdx.x >> 5);
  const int tn = (int)blockIdx.x * 8 + wave;
  const int tm = (int)blockIdx.y;
  if (tn * 16 >= N) return;  // wave-uniform exit
  v8f acc = {};
  const _Float16* Ab = A + (size_t)tm * 16 * K;
  const _Float16* Bb = B + tn * 16;
  for (int k = 0; k < K; k += 32) {
    const v16h a = load_a_frag(Ab + k, K);
    const v16h b = load_b_frag(Bb + (size_t)k * N, N);
    acc = __builtin_amdgcn_wmma_f32_16x16x32_f16(false, a, false, b, (short)0, acc, false, false);
  }
  const int lane = lane_id();
  const int col = tn * 16 + (lane & 15);
  const int row0 = tm * 16 + ((lane >> 4) << 3);
#pragma unroll
  for (int r = 0; r < 8; ++r) C[(size_t)(row0 + r) * N + col] = (_Float16)acc[r];
}

// ---------------------------------------------------------------------------
// Kernel 6: fused output head + concat:
//   r[b,o,n] = Wcomb[o,:] . [gsum;x1;x2][b,:,n] + bias_comb[o]
//   cat[b,t, 0:N]=r, [N:2N]=y, [2N:3N]=ymask
// ---------------------------------------------------------------------------
__global__ __launch_bounds__(256) void proj_concat_kernel(
    const _Float16* gsum, const _Float16* x1s, const _Float16* x2s,
    const float* Wcomb, const float* biasc,
    const float* y, const float* ymask, float* cat) {
  const size_t idx = (size_t)blockIdx.x * 256 + threadIdx.x;
  if (idx >= (size_t)BB * TT * 3 * NN) return;
  const int j = (int)(idx % (3 * NN));
  const int row = (int)(idx / (3 * NN));  // b*62 + o
  if (j < NN) {
    const int b = row / TT, o = row - b * TT;
    float acc = biasc[o];
    const float* wr = Wcomb + o * 192;
    const size_t base = (size_t)(b * 64) * NPAD + j;
    for (int i = 0; i < 64; ++i) acc += wr[i]       * (float)gsum[base + (size_t)i * NPAD];
    for (int i = 0; i < 64; ++i) acc += wr[64 + i]  * (float)x1s[base + (size_t)i * NPAD];
    for (int i = 0; i < 64; ++i) acc += wr[128 + i] * (float)x2s[base + (size_t)i * NPAD];
    cat[idx] = acc;
  } else if (j < 2 * NN) {
    cat[idx] = y[(size_t)row * NN + (j - NN)];
  } else {
    cat[idx] = ymask[(size_t)row * NN + (j - 2 * NN)];
  }
}

// ---------------------------------------------------------------------------
// Kernels 7/8: final MLP (sub-GFLOP, plain VALU)
// ---------------------------------------------------------------------------
__global__ __launch_bounds__(256) void fin1_kernel(const float* cat, const float* w,
                                                   const float* b, float* q) {
  const int idx = (int)(blockIdx.x * 256 + threadIdx.x);
  if (idx >= BB * TT * 128) return;
  const int row = idx >> 7, o = idx & 127;
  float acc = b[o];
  const float* wr = w + (size_t)o * (3 * NN);
  const float* cr = cat + (size_t)row * (3 * NN);
  for (int k = 0; k < 3 * NN; ++k) acc += wr[k] * cr[k];
  q[idx] = acc;
}

__global__ __launch_bounds__(256) void fin2_kernel(const float* q, const float* w,
                                                   const float* b, float* out) {
  const int idx = (int)(blockIdx.x * 256 + threadIdx.x);
  if (idx >= BB * TT * NN) return;
  const int row = idx / NN, n = idx - row * NN;
  float acc = b[n];
  const float* qr = q + row * 128;
  const float* wr = w + n * 128;
  for (int k = 0; k < 128; ++k) acc += qr[k] * wr[k];
  out[idx] = acc;
}

// ---------------------------------------------------------------------------
// Workspace layout (bytes)
// ---------------------------------------------------------------------------
constexpr size_t HSEQ_E     = (size_t)MSEQ * TT * HH;          // 33,585,152 f16
constexpr size_t OFF_HFW    = 0;
constexpr size_t OFF_HBW    = OFF_HFW + HSEQ_E * 2;
constexpr size_t OFF_ADP    = OFF_HBW + HSEQ_E * 2;
constexpr size_t OFF_GSUM   = OFF_ADP + (size_t)NPAD * NPAD * 2;
constexpr size_t OFF_X1     = OFF_GSUM + (size_t)MROW * NPAD * 2;
constexpr size_t OFF_X2     = OFF_X1 + (size_t)MROW * NPAD * 2;
constexpr size_t OFF_WHT_FW = OFF_X2 + (size_t)MROW * NPAD * 2;
constexpr size_t OFF_WHT_BW = OFF_WHT_FW + (size_t)HH * G4 * 2;
constexpr size_t OFF_VEC_FW = OFF_WHT_BW + (size_t)HH * G4 * 2;
constexpr size_t OFF_VEC_BW = OFF_VEC_FW + 768 * 4;
constexpr size_t OFF_E      = OFF_VEC_BW + 768 * 4;
constexpr size_t OFF_WCOMB  = OFF_E + (size_t)TT * CC * 4;
constexpr size_t OFF_BIASC  = OFF_WCOMB + (size_t)TT * 192 * 4;
constexpr size_t OFF_CAT    = OFF_BIASC + 64 * 4;
constexpr size_t OFF_Q      = OFF_CAT + (size_t)BB * TT * 3 * NN * 4;

extern "C" void kernel_launch(void* const* d_in, const int* in_sizes, int n_in,
                              void* d_out, int out_size, void* d_ws, size_t ws_size,
                              hipStream_t stream) {
  const float* it_fw  = (const float*)d_in[0];
  const float* mk_fw  = (const float*)d_in[1];
  const float* it_bw  = (const float*)d_in[2];
  const float* mk_bw  = (const float*)d_in[3];
  const float* y      = (const float*)d_in[4];
  const float* ymask  = (const float*)d_in[5];
  const float* sw     = (const float*)d_in[6];
  const float* sb     = (const float*)d_in[7];
  const float* cw     = (const float*)d_in[8];
  const float* cb     = (const float*)d_in[9];
  const float* wih_fw = (const float*)d_in[10];
  const float* whh_fw = (const float*)d_in[11];
  const float* bih_fw = (const float*)d_in[12];
  const float* bhh_fw = (const float*)d_in[13];
  const float* wih_bw = (const float*)d_in[14];
  const float* whh_bw = (const float*)d_in[15];
  const float* bih_bw = (const float*)d_in[16];
  const float* bhh_bw = (const float*)d_in[17];
  const float* nv1    = (const float*)d_in[18];
  const float* nv2    = (const float*)d_in[19];
  const float* gconv_w = (const float*)d_in[20];
  const float* gconv_b = (const float*)d_in[21];
  const float* end1_w = (const float*)d_in[22];
  const float* end1_b = (const float*)d_in[23];
  const float* end2_w = (const float*)d_in[24];
  const float* end2_b = (const float*)d_in[25];
  const float* lin_w  = (const float*)d_in[26];
  const float* lin_b  = (const float*)d_in[27];
  const float* fin1_w = (const float*)d_in[28];
  const float* fin1_b = (const float*)d_in[29];
  const float* fin2_w = (const float*)d_in[30];
  const float* fin2_b = (const float*)d_in[31];

  char* ws = (char*)d_ws;
  _Float16* hfw    = (_Float16*)(ws + OFF_HFW);
  _Float16* hbw    = (_Float16*)(ws + OFF_HBW);
  _Float16* adp    = (_Float16*)(ws + OFF_ADP);
  _Float16* gsum   = (_Float16*)(ws + OFF_GSUM);
  _Float16* x1s    = (_Float16*)(ws + OFF_X1);
  _Float16* x2s    = (_Float16*)(ws + OFF_X2);
  _Float16* whT_fw = (_Float16*)(ws + OFF_WHT_FW);
  _Float16* whT_bw = (_Float16*)(ws + OFF_WHT_BW);
  float* vec_fw = (float*)(ws + OFF_VEC_FW);
  float* vec_bw = (float*)(ws + OFF_VEC_BW);
  float* E      = (float*)(ws + OFF_E);
  float* Wcomb  = (float*)(ws + OFF_WCOMB);
  float* biasc  = (float*)(ws + OFF_BIASC);
  float* cat    = (float*)(ws + OFF_CAT);
  float* q      = (float*)(ws + OFF_Q);

  prep_kernel<<<1, 256, 0, stream>>>(
      sw, sb, cw, cb,
      wih_fw, whh_fw, bih_fw, bhh_fw,
      wih_bw, whh_bw, bih_bw, bhh_bw,
      gconv_w, gconv_b, end1_w, end1_b, end2_w, end2_b, lin_w, lin_b,
      vec_fw, vec_bw, whT_fw, whT_bw, E, Wcomb, biasc);

  adp_kernel<<<NPAD, 256, 0, stream>>>(nv1, nv2, adp);

  lstm_kernel<<<dim3(MSEQ / 16, 2), 256, 0, stream>>>(
      it_fw, mk_fw, it_bw, mk_bw, whT_fw, whT_bw, vec_fw, vec_bw, hfw, hbw);

  combine_kernel<<<(MROW * NPAD) / 256, 256, 0, stream>>>(hfw, hbw, lin_w, gsum);

  const dim3 ggrid((NPAD / 16 + 7) / 8, MROW / 16);
  gemm_f16<<<ggrid, 256, 0, stream>>>(gsum, adp, x1s, MROW, NPAD, NPAD);
  gemm_f16<<<ggrid, 256, 0, stream>>>(x1s, adp, x2s, MROW, NPAD, NPAD);

  const size_t cat_total = (size_t)BB * TT * 3 * NN;
  proj_concat_kernel<<<(unsigned)((cat_total + 255) / 256), 256, 0, stream>>>(
      gsum, x1s, x2s, Wcomb, biasc, y, ymask, cat);

  fin1_kernel<<<(BB * TT * 128 + 255) / 256, 256, 0, stream>>>(cat, fin1_w, fin1_b, q);
  fin2_kernel<<<(BB * TT * NN + 255) / 256, 256, 0, stream>>>(q, fin2_w, fin2_b,
                                                              (float*)d_out);
}
